// SemanticSimModel_5463198400969
// MI455X (gfx1250) — matrix operand order repaired and checked by reading, not compiled, not run on hardware
//
#include <hip/hip_runtime.h>

typedef __attribute__((ext_vector_type(16))) __bf16 v16bf;
typedef __attribute__((ext_vector_type(8)))  __bf16 v8bf;
typedef __attribute__((ext_vector_type(8)))  float  v8f;
typedef __attribute__((ext_vector_type(4)))  unsigned int u32x4;
typedef __attribute__((ext_vector_type(8)))  unsigned int u32x8;

#define VSZ 32000
#define ESZ 300
#define EP  320      // K padded to multiple of 32
#define HSZ 512
#define G4  2048     // 4*H
#define GCAT 4096    // fwd+bwd gate columns
#define BSZ 64
#define TSZ 128
#define BT  8192     // B*T
#define CSZ 256
#define MSZ 512

__device__ __forceinline__ float sigm(float x) { return 1.0f / (1.0f + __expf(-x)); }
__device__ __forceinline__ __bf16 f2bf(float f) { return (__bf16)f; }
__device__ __forceinline__ float bf2f(__bf16 h) { return (float)h; }

__device__ __forceinline__ v8f wmma_bf16(v16bf a, v16bf b, v8f c) {
  // D(f32 16x16) = A(bf16 16x32) * B(bf16 32x16) + C
  return __builtin_amdgcn_wmma_f32_16x16x32_bf16(false, a, false, b, (short)0, c,
                                                 false, false);
}

// A fragment (16x32 bf16). rowp = this lane's row base (row-major, lane owns row lane&15).
// Lanes 0-15 hold K {k..k+15}; lanes 16-31 hold K {k+16..k+31}.
__device__ __forceinline__ v16bf load_a(const __bf16* rowp, int k) {
  int kh = ((threadIdx.x & 31) >> 4) << 4;  // 0 or 16
  v8bf lo = *(const v8bf*)(rowp + k + kh);
  v8bf hi = *(const v8bf*)(rowp + k + 8 + kh);
  v16bf a;
#pragma unroll
  for (int i = 0; i < 8; ++i) { a[i] = lo[i]; a[8 + i] = hi[i]; }
  return a;
}

// B fragment (32x16 bf16), B[k][n] = W[n][k] with W row-major [N, ldk] (k contiguous).
// Lane n (0-15): K k..k+15 packed 2/VGPR; lane n+16: K k+16..k+31.
__device__ __forceinline__ v16bf load_bT(const __bf16* W, int ldk, int n0, int k) {
  int lane = threadIdx.x & 31;
  int n = lane & 15;
  int kb = (lane >> 4) << 4;  // 0 or 16
  return *(const v16bf*)(W + (size_t)(n0 + n) * ldk + k + kb);
}

// Async copy of `bytes16` 16-byte chunks from global to LDS (per-block cooperative).
__device__ __forceinline__ void async_stage(const void* gsrc, void* ldst, int chunks,
                                            int nthreads) {
  unsigned ldsb = (unsigned)(uintptr_t)ldst;  // low 32 bits of LDS flat addr = LDS offset
  const char* g = (const char*)gsrc;
  for (int i = threadIdx.x; i < chunks; i += nthreads) {
    unsigned dst = ldsb + (unsigned)i * 16u;
    const char* src = g + (size_t)i * 16;
    asm volatile("global_load_async_to_lds_b128 %0, %1, off" ::"v"(dst), "v"(src)
                 : "memory");
  }
  asm volatile("s_wait_asynccnt 0x0" ::: "memory");
}

// ---------------- weight conversion ----------------

__global__ void k_pad_rows(const float* __restrict__ src, __bf16* __restrict__ dst,
                           int R, int K, int KP) {
  int stride = gridDim.x * blockDim.x;
  int total = R * KP;
  for (int i = blockIdx.x * blockDim.x + threadIdx.x; i < total; i += stride) {
    int r = i / KP, k = i - r * KP;
    dst[i] = f2bf(k < K ? src[(size_t)r * K + k] : 0.0f);
  }
}

__global__ void k_cvt(const float* __restrict__ src, __bf16* __restrict__ dst, int n) {
  int stride = gridDim.x * blockDim.x;
  for (int i = blockIdx.x * blockDim.x + threadIdx.x; i < n; i += stride)
    dst[i] = f2bf(src[i]);
}

// ---------------- embedding gather + input projection GEMM ----------------
// xw[row, 0:2048] = emb[tok[row]] @ Wih_f^T + b_f ; [2048:4096] same with Wih_b,b_b
// B slab (64 cols x 320 K, 40KB) staged per block via async-to-LDS.
__global__ void k_xw(const int* __restrict__ s1, const int* __restrict__ s2,
                     const __bf16* __restrict__ emb_pad, const __bf16* __restrict__ wih_cat,
                     const float* __restrict__ b_f, const float* __restrict__ b_b,
                     __bf16* __restrict__ xw) {
  __shared__ __bf16 bsm[64 * EP];  // 40 KB
  int blk = blockIdx.x;
  int bn = blk & 63, bm = blk >> 6;  // 64 col-blocks(64) x 128 row-blocks(128)
  int w = threadIdx.x >> 5, lane = threadIdx.x & 31;
  int rowbase = bm * 128 + w * 16;
  int colbase = bn * 64;

  async_stage(wih_cat + (size_t)colbase * EP, bsm, (64 * EP * 2) / 16, 256);
  __syncthreads();

  int arow = rowbase + (lane & 15);
  int tok = (arow < BT) ? s1[arow] : s2[arow - BT];
  const __bf16* ap = emb_pad + (size_t)tok * EP;
  v8f acc[4] = {};
  v16bf a = load_a(ap, 0);
#pragma unroll 1
  for (int kk = 0; kk < 9; ++kk) {
    int k = kk * 32;
    v16bf an = load_a(ap, k + 32);  // pipeline next A
#pragma unroll
    for (int nt = 0; nt < 4; ++nt) {
      v16bf b = load_bT(bsm, EP, nt * 16, k);
      acc[nt] = wmma_bf16(a, b, acc[nt]);
    }
    a = an;
  }
#pragma unroll
  for (int nt = 0; nt < 4; ++nt) {
    v16bf b = load_bT(bsm, EP, nt * 16, EP - 32);
    acc[nt] = wmma_bf16(a, b, acc[nt]);
  }

  int n = lane & 15, hi8 = (lane >> 4) * 8;
#pragma unroll
  for (int nt = 0; nt < 4; ++nt) {
    int col = colbase + nt * 16 + n;
    float bias = (col < G4) ? b_f[col] : b_b[col - G4];
#pragma unroll
    for (int v = 0; v < 8; ++v) {
      int row = rowbase + v + hi8;
      xw[(size_t)row * GCAT + col] = f2bf(acc[nt][v] + bias);
    }
  }
}

// ---------------- persistent recurrent LSTM (4 independent chains) ----------------
// One workgroup per chain. h double-buffered bf16 in LDS, c fp32 in LDS.
// 16 waves; wave w owns hidden column tiles j = w and w+16 (each with its 4 gate
// tiles i/f/g/o across all 4 batch row-tiles) -> gate elementwise needs no sync.
// Whh B-fragments double-buffered in registers to overlap L2 stream with WMMA.
__global__ void k_lstm(const __bf16* __restrict__ xw, const __bf16* __restrict__ whh_f,
                       const __bf16* __restrict__ whh_b,
                       const float* __restrict__ s1h0, const float* __restrict__ s1c0,
                       const float* __restrict__ s2h0, const float* __restrict__ s2c0,
                       const int* __restrict__ len1, const int* __restrict__ len2,
                       __bf16* __restrict__ hs) {
  extern __shared__ char smem[];
  float*  c_s   = (float*)smem;              // 131072 B
  __bf16* hbuf0 = (__bf16*)(smem + 131072);  // 65536 B
  __bf16* hbuf1 = (__bf16*)(smem + 196608);  // 65536 B
  int*    lens_s = (int*)(smem + 262144);    // 256 B

  int chain = blockIdx.x;
  int seq = chain >> 1, dir = chain & 1;
  const float* h0 = (seq ? s2h0 : s1h0) + (size_t)dir * BSZ * HSZ;
  const float* c0 = (seq ? s2c0 : s1c0) + (size_t)dir * BSZ * HSZ;
  const int* lens = seq ? len2 : len1;
  const __bf16* Whh = dir ? whh_b : whh_f;
  const __bf16* xwbase = xw + (size_t)seq * BT * GCAT + dir * G4;
  __bf16* hsbase = hs + (size_t)seq * BSZ * TSZ * 1024 + dir * HSZ;

  int tid = threadIdx.x;
  int w = tid >> 5, lane = tid & 31;
  for (int i = tid; i < BSZ * HSZ; i += 512) {
    c_s[i] = c0[i];
    hbuf0[i] = f2bf(h0[i]);
  }
  if (tid < BSZ) lens_s[tid] = lens[tid];
  __syncthreads();

  for (int t = 0; t < TSZ; ++t) {
    const __bf16* cur = (t & 1) ? hbuf1 : hbuf0;
    __bf16* nxt = (t & 1) ? hbuf0 : hbuf1;
    for (int pp = 0; pp < 2; ++pp) {
      int j = w + pp * 16;  // hidden col tile 0..31
      v8f acc[4][4] = {};   // [row tile][gate]
      v16bf bfr[4];
#pragma unroll
      for (int g = 0; g < 4; ++g) bfr[g] = load_bT(Whh, HSZ, g * HSZ + j * 16, 0);
#pragma unroll 1
      for (int kk = 0; kk < 15; ++kk) {
        int k = kk * 32;
        __builtin_prefetch(Whh + (size_t)(j * 16) * HSZ + k + 96, 0, 1);
        v16bf bnx[4];
#pragma unroll
        for (int g = 0; g < 4; ++g)
          bnx[g] = load_bT(Whh, HSZ, g * HSZ + j * 16, k + 32);  // pipeline next B
        v16bf afr[4];
#pragma unroll
        for (int r = 0; r < 4; ++r)
          afr[r] = load_a(cur + (size_t)(r * 16 + (lane & 15)) * HSZ, k);
#pragma unroll
        for (int g = 0; g < 4; ++g) {
#pragma unroll
          for (int r = 0; r < 4; ++r) acc[r][g] = wmma_bf16(afr[r], bfr[g], acc[r][g]);
        }
#pragma unroll
        for (int g = 0; g < 4; ++g) bfr[g] = bnx[g];
      }
      {  // final K step (k = 480)
        v16bf afr[4];
#pragma unroll
        for (int r = 0; r < 4; ++r)
          afr[r] = load_a(cur + (size_t)(r * 16 + (lane & 15)) * HSZ, HSZ - 32);
#pragma unroll
        for (int g = 0; g < 4; ++g) {
#pragma unroll
          for (int r = 0; r < 4; ++r) acc[r][g] = wmma_bf16(afr[r], bfr[g], acc[r][g]);
        }
      }
      // gates: lane element (m=v+8*hi, n)
      int n = lane & 15, hi8 = (lane >> 4) * 8;
      int jc = j * 16 + n;  // hidden unit index
#pragma unroll
      for (int r = 0; r < 4; ++r) {
#pragma unroll
        for (int v = 0; v < 8; ++v) {
          int b = r * 16 + v + hi8;
          int len = lens_s[b];
          int p = dir ? ((t < len) ? (len - 1 - t) : t) : t;  // reversed-within-valid
          const __bf16* xr = xwbase + ((size_t)b * TSZ + p) * GCAT;
          float gi = acc[r][0][v] + bf2f(xr[jc]);
          float gf = acc[r][1][v] + bf2f(xr[HSZ + jc]);
          float gg = acc[r][2][v] + bf2f(xr[2 * HSZ + jc]);
          float go = acc[r][3][v] + bf2f(xr[3 * HSZ + jc]);
          float cold = c_s[b * HSZ + jc];
          float cn = sigm(gf) * cold + sigm(gi) * tanhf(gg);
          float hn = sigm(go) * tanhf(cn);
          c_s[b * HSZ + jc] = cn;
          nxt[b * HSZ + jc] = f2bf(hn);
          hsbase[((size_t)b * TSZ + p) * 1024 + jc] = f2bf(hn);  // un-reversed store
        }
      }
    }
    __syncthreads();
  }
}

// ---------------- attention projection: P = tanh(Hout @ S1W^T) ----------------
// S1W slab (64 rows x 1024 K bf16 = 128 KB) DMA'd to LDS by the Tensor Data Mover.
__global__ void k_attn(const __bf16* __restrict__ hs, const __bf16* __restrict__ s1w_bf,
                       __bf16* __restrict__ P) {
  extern __shared__ char smem[];
  __bf16* bsm = (__bf16*)smem;  // 64 * 1024 bf16 = 128 KB
  int blk = blockIdx.x;
  int cb = blk & 3, rb = blk >> 2;  // 4 col-blocks(64) x 128 row-blocks(128)
  int w = threadIdx.x >> 5, lane = threadIdx.x & 31;
  int rowbase = rb * 128 + w * 16;
  int colbase = cb * 64;

  if (threadIdx.x < 32) {  // wave 0 issues one TDM descriptor for the whole slab
    unsigned lds_addr = (unsigned)(uintptr_t)bsm;
    unsigned long long ga =
        (unsigned long long)(uintptr_t)(s1w_bf + (size_t)colbase * 1024);
    u32x4 g0;
    g0[0] = 1u;                                   // count=1 (valid user D#)
    g0[1] = lds_addr;                             // LDS byte address
    g0[2] = (unsigned)(ga & 0xFFFFFFFFu);         // global_addr[31:0]
    g0[3] = (unsigned)((ga >> 32) & 0x01FFFFFFu)  // global_addr[56:32]
            | (2u << 30);                         // type=2 (image)
    u32x8 g1;
    g1[0] = (1u << 16);     // data_size=1 (2 bytes/elem)
    g1[1] = (1024u << 16);  // tensor_dim0[15:0] in [31:16]
    g1[2] = (64u << 16);    // tensor_dim0 hi=0 | tensor_dim1[15:0]=64
    g1[3] = (1024u << 16);  // tensor_dim1 hi=0 | tile_dim0=1024
    g1[4] = 64u;            // tile_dim1=64, tile_dim2=0
    g1[5] = 1024u;          // tensor_dim0_stride[31:0] (elements)
    g1[6] = 0u;             // stride hi | tensor_dim1_stride lo
    g1[7] = 0u;
    asm volatile("tensor_load_to_lds %0, %1" ::"s"(g0), "s"(g1) : "memory");
    __builtin_amdgcn_s_wait_tensorcnt(0);
  }
  __syncthreads();

  const __bf16* ap = hs + (size_t)(rowbase + (lane & 15)) * 1024;
  v8f acc[4] = {};
  v16bf a = load_a(ap, 0);
#pragma unroll 1
  for (int kk = 0; kk < 31; ++kk) {
    int k = kk * 32;
    v16bf an = load_a(ap, k + 32);  // pipeline next A
#pragma unroll
    for (int nt = 0; nt < 4; ++nt) {
      v16bf b = load_bT(bsm, 1024, nt * 16, k);
      acc[nt] = wmma_bf16(a, b, acc[nt]);
    }
    a = an;
  }
#pragma unroll
  for (int nt = 0; nt < 4; ++nt) {
    v16bf b = load_bT(bsm, 1024, nt * 16, 1024 - 32);
    acc[nt] = wmma_bf16(a, b, acc[nt]);
  }

  int n = lane & 15, hi8 = (lane >> 4) * 8;
#pragma unroll
  for (int nt = 0; nt < 4; ++nt) {
    int col = colbase + nt * 16 + n;
#pragma unroll
    for (int v = 0; v < 8; ++v) {
      int row = rowbase + v + hi8;
      P[(size_t)row * CSZ + col] = f2bf(tanhf(acc[nt][v]));
    }
  }
}

// ---------------- masked softmax pooling ----------------
__global__ void k_pool(const __bf16* __restrict__ P, const __bf16* __restrict__ hs,
                       const float* __restrict__ S2W, const int* __restrict__ len1,
                       const int* __restrict__ len2, float* __restrict__ rbuf) {
  __shared__ float s2w[CSZ];
  __shared__ float sc[TSZ];
  __shared__ float red[TSZ];
  int sb = blockIdx.x, seq = sb >> 6, b = sb & 63;
  int tid = threadIdx.x;  // 128
  for (int c = tid; c < CSZ; c += 128) s2w[c] = S2W[c];
  __syncthreads();
  int len = (seq ? len2 : len1)[b];
  const __bf16* Prow = P + ((size_t)(seq * BSZ + b) * TSZ + tid) * CSZ;
  float s = 0.0f;
  for (int c = 0; c < CSZ; ++c) s += bf2f(Prow[c]) * s2w[c];
  float ms = (tid < len) ? s : -1e9f;
  sc[tid] = ms;
  red[tid] = ms;
  __syncthreads();
  for (int st = 64; st > 0; st >>= 1) {
    if (tid < st) red[tid] = fmaxf(red[tid], red[tid + st]);
    __syncthreads();
  }
  float mx = red[0];
  __syncthreads();
  float e = __expf(sc[tid] - mx);
  red[tid] = e;
  __syncthreads();
  for (int st = 64; st > 0; st >>= 1) {
    if (tid < st) red[tid] += red[tid + st];
    __syncthreads();
  }
  float inv = 1.0f / red[0];
  __syncthreads();
  sc[tid] = e * inv;
  __syncthreads();
  const __bf16* Hrow = hs + (size_t)(seq * BSZ + b) * TSZ * 1024;
  float* rout = rbuf + (size_t)(seq * BSZ + b) * 1024;
  for (int d = tid; d < 1024; d += 128) {
    float acc = 0.0f;
    for (int t = 0; t < TSZ; ++t) acc += sc[t] * bf2f(Hrow[(size_t)t * 1024 + d]);
    rout[d] = acc;
  }
}

// ---------------- merge + MLP + sigmoid head ----------------
__global__ void k_head(const float* __restrict__ rbuf, const float* __restrict__ mlpb,
                       const float* __restrict__ outW, const float* __restrict__ outb,
                       const __bf16* __restrict__ mlpw_bf, __bf16* __restrict__ merged,
                       float* __restrict__ outmlp, float* __restrict__ out) {
  int tid = threadIdx.x;  // 256
  // phase A: merged = [r1+r2, (r1-r2)^2]  (64 x 2048, bf16)
  for (int i = tid; i < BSZ * G4; i += 256) {
    int b = i >> 11, d = i & 2047;
    float v;
    if (d < 1024) {
      v = rbuf[(size_t)b * 1024 + d] + rbuf[(size_t)(BSZ + b) * 1024 + d];
    } else {
      float t1 = rbuf[(size_t)b * 1024 + d - 1024];
      float t2 = rbuf[(size_t)(BSZ + b) * 1024 + d - 1024];
      float df = t1 - t2;
      v = df * df;
    }
    merged[i] = f2bf(v);
  }
  __threadfence();
  __syncthreads();
  // phase B: outmlp = merged @ mlpW^T + mlpb   (64x512, K=2048)
  int w = tid >> 5, lane = tid & 31;
  v8f acc[4][4] = {};
  for (int kk = 0; kk < 64; ++kk) {
    int k = kk * 32;
    v16bf afr[4];
#pragma unroll
    for (int r = 0; r < 4; ++r)
      afr[r] = load_a(merged + (size_t)(r * 16 + (lane & 15)) * G4, k);
#pragma unroll
    for (int ct = 0; ct < 4; ++ct) {
      v16bf bfr = load_bT(mlpw_bf, G4, w * 64 + ct * 16, k);
#pragma unroll
      for (int r = 0; r < 4; ++r) acc[r][ct] = wmma_bf16(afr[r], bfr, acc[r][ct]);
    }
  }
  int n = lane & 15, hi8 = (lane >> 4) * 8;
#pragma unroll
  for (int r = 0; r < 4; ++r) {
#pragma unroll
    for (int ct = 0; ct < 4; ++ct) {
      int col = w * 64 + ct * 16 + n;
#pragma unroll
      for (int v = 0; v < 8; ++v) {
        int row = r * 16 + v + hi8;
        outmlp[(size_t)row * MSZ + col] = acc[r][ct][v] + mlpb[col];
      }
    }
  }
  __threadfence();
  __syncthreads();
  // phase C: sigmoid(outmlp @ outW^T + outb)
  if (tid < BSZ) {
    float s = 0.0f;
    for (int m = 0; m < MSZ; ++m) s += outmlp[(size_t)tid * MSZ + m] * outW[m];
    out[tid] = sigm(s + outb[0]);
  }
}

// ---------------- launcher ----------------
extern "C" void kernel_launch(void* const* d_in, const int* in_sizes, int n_in,
                              void* d_out, int out_size, void* d_ws, size_t ws_size,
                              hipStream_t stream) {
  const int*   s1    = (const int*)d_in[0];
  const int*   s2    = (const int*)d_in[1];
  const int*   len1  = (const int*)d_in[2];
  const int*   len2  = (const int*)d_in[3];
  const float* s1h0  = (const float*)d_in[4];
  const float* s1c0  = (const float*)d_in[5];
  const float* s2h0  = (const float*)d_in[6];
  const float* s2c0  = (const float*)d_in[7];
  const float* emb   = (const float*)d_in[8];
  const float* Wih_f = (const float*)d_in[9];
  const float* Whh_f = (const float*)d_in[10];
  const float* b_f   = (const float*)d_in[11];
  const float* Wih_b = (const float*)d_in[12];
  const float* Whh_b = (const float*)d_in[13];
  const float* b_b   = (const float*)d_in[14];
  const float* S1W   = (const float*)d_in[15];
  const float* S2W   = (const float*)d_in[16];
  const float* mlpW  = (const float*)d_in[17];
  const float* mlpb  = (const float*)d_in[18];
  const float* outW  = (const float*)d_in[19];
  const float* outb  = (const float*)d_in[20];

  char* ws = (char*)d_ws;
  size_t off = 0;
  auto take = [&](size_t bytes) {
    char* p = ws + off;
    off += (bytes + 255) & ~(size_t)255;
    return p;
  };
  __bf16* emb_pad = (__bf16*)take((size_t)VSZ * EP * 2);
  __bf16* wih_cat = (__bf16*)take((size_t)GCAT * EP * 2);
  __bf16* whhf_bf = (__bf16*)take((size_t)G4 * HSZ * 2);
  __bf16* whhb_bf = (__bf16*)take((size_t)G4 * HSZ * 2);
  __bf16* s1w_bf  = (__bf16*)take((size_t)CSZ * 1024 * 2);
  __bf16* mlpw_bf = (__bf16*)take((size_t)MSZ * G4 * 2);
  __bf16* xwbuf   = (__bf16*)take((size_t)2 * BT * GCAT * 2);
  __bf16* hsbuf   = (__bf16*)take((size_t)2 * BT * 1024 * 2);
  __bf16* Pbuf    = (__bf16*)take((size_t)2 * BT * CSZ * 2);
  float*  rbuf    = (float*)take((size_t)2 * BSZ * 1024 * 4);
  __bf16* merged  = (__bf16*)take((size_t)BSZ * G4 * 2);
  float*  outmlp  = (float*)take((size_t)BSZ * MSZ * 4);

  // weight conversion / padding (K 300 -> 320 with zeros)
  k_pad_rows<<<2048, 256, 0, stream>>>(emb, emb_pad, VSZ, ESZ, EP);
  k_pad_rows<<<512, 256, 0, stream>>>(Wih_f, wih_cat, G4, ESZ, EP);
  k_pad_rows<<<512, 256, 0, stream>>>(Wih_b, wih_cat + (size_t)G4 * EP, G4, ESZ, EP);
  k_cvt<<<512, 256, 0, stream>>>(Whh_f, whhf_bf, G4 * HSZ);
  k_cvt<<<512, 256, 0, stream>>>(Whh_b, whhb_bf, G4 * HSZ);
  k_cvt<<<256, 256, 0, stream>>>(S1W, s1w_bf, CSZ * 1024);
  k_cvt<<<512, 256, 0, stream>>>(mlpW, mlpw_bf, MSZ * G4);

  // fused embedding gather + dual-direction input projection (async B staging)
  k_xw<<<8192, 256, 0, stream>>>(s1, s2, emb_pad, wih_cat, b_f, b_b, xwbuf);

  // 4 persistent recurrent chains (256 KB + 256 B dynamic LDS each)
  k_lstm<<<4, 512, 262400, stream>>>(xwbuf, whhf_bf, whhb_bf, s1h0, s1c0, s2h0, s2c0,
                                     len1, len2, hsbuf);

  // attention projection (TDM-staged B slab, 128 KB LDS) + pooling
  k_attn<<<512, 256, 131072, stream>>>(hsbuf, s1w_bf, Pbuf);
  k_pool<<<128, 128, 0, stream>>>(Pbuf, hsbuf, S2W, len1, len2, rbuf);

  // merge + MLP + sigmoid
  k_head<<<1, 256, 0, stream>>>(rbuf, mlpb, outW, outb, mlpw_bf, merged, outmlp,
                                (float*)d_out);
}
